// MultiHeadAttention_13297218748655
// MI455X (gfx1250) — matrix operand order repaired and checked by reading, not compiled, not run on hardware
//
#include <hip/hip_runtime.h>

// MHA forward, bf16 WMMA datapath (f32 accumulate), MI455X / gfx1250.
// B=2 T=2048 C=1024 H=16 D=64.
// Data staging uses CDNA5 async global->LDS copies (ASYNCcnt) + split barriers.

#define EMB   1024
#define NHEAD 16
#define HEADD 64
#define BSZ   2
#define SEQ   2048
#define MTOK  (BSZ * SEQ)   // 4096 token rows
#define N_QKV (3 * EMB)     // 3072

typedef __attribute__((ext_vector_type(16))) __bf16 v16bf;
typedef __attribute__((ext_vector_type(8)))  float  v8f;

union ABu { v16bf v; unsigned u[8]; };

__device__ inline __bf16 f2bf(float f) {
  union { float f; unsigned u; } a; a.f = f;
  unsigned r = a.u + 0x7FFFu + ((a.u >> 16) & 1u);   // round-to-nearest-even
  union { unsigned short s; __bf16 b; } o; o.s = (unsigned short)(r >> 16);
  return o.b;
}

// CDNA5 async copy: 16B global -> LDS, tracked by ASYNCcnt.
// lds_off = LDS byte offset (low 32 bits of flat address of a __shared__ object).
__device__ inline void async_ld_b128(unsigned lds_off, const void* gaddr) {
  asm volatile("global_load_async_to_lds_b128 %0, %1, off"
               :: "v"(lds_off), "v"(gaddr) : "memory");
}
__device__ inline void wait_async0() {
  asm volatile("s_wait_asynccnt 0x0" ::: "memory");
}
__device__ inline unsigned lds_off32(const void* p) {
  return (unsigned)(unsigned long long)p;   // LDS aperture: addr[31:0] = LDS offset
}

__global__ void cvt_f32_bf16(const float* __restrict__ s, __bf16* __restrict__ d, int n) {
  int i = blockIdx.x * 256 + threadIdx.x;
  if (i < n) d[i] = f2bf(s[i]);
}

// A fragment: 16x32 bf16 from [row][ld] (K contiguous per row).
// ISA: lanes 0-15 row M=lane, K{0..7} in v0-3, K{16..23} in v4-7; lanes 16-31 K+8.
__device__ inline v16bf load_a_frag(const __bf16* base, int ld) {
  int lane = threadIdx.x & 31;
  int row = lane & 15, half = lane >> 4;
  ABu r;
  const unsigned* p = (const unsigned*)(base + row * ld);
#pragma unroll
  for (int j = 0; j < 4; ++j) {
    r.u[j]     = p[half * 4 + j];
    r.u[4 + j] = p[8 + half * 4 + j];
  }
  return r.v;
}

// B fragment: 32(K) x 16(N) bf16 from [col][ld] (K contiguous per column).
__device__ inline v16bf load_b_frag(const __bf16* base, int ld) {
  int lane = threadIdx.x & 31;
  int col = lane & 15, kh = lane >> 4;
  ABu r;
  const unsigned* p = (const unsigned*)(base + col * ld);
#pragma unroll
  for (int j = 0; j < 4; ++j) {
    r.u[j]     = p[kh * 4 + j];
    r.u[4 + j] = p[8 + kh * 4 + j];
  }
  return r.v;
}

__device__ inline v8f wmma_bf16(v16bf a, v16bf b, v8f c) {
  return __builtin_amdgcn_wmma_f32_16x16x32_bf16(false, a, false, b, (short)0, c, false, false);
}

// C[M,N] = A[M,K] (row-major bf16) * B[N,K]^T (row-major bf16) + bias[N]
// mode 0: fp32 out[M,N].  mode 1: scatter to q/k [B,H,T,D] and vT [B,H,D,T].
__global__ void __launch_bounds__(128)
gemm_bf16_nt(const __bf16* __restrict__ Ag, const __bf16* __restrict__ Bg,
             const float* __restrict__ bias, float* __restrict__ outF,
             __bf16* __restrict__ qO, __bf16* __restrict__ kO, __bf16* __restrict__ vTO,
             int Ndim, int Kdim, int mode) {
  __shared__ __bf16 At[64 * 40];   // 64 rows x 32 K (pad ld=40)
  __shared__ __bf16 Bt[64 * 40];
  const int tid  = threadIdx.x;
  const int wave = tid >> 5;
  const int lane = tid & 31;
  const int bm = blockIdx.y * 64;
  const int bn = blockIdx.x * 64;

  v8f acc[4] = {};

  const int kIters = Kdim >> 5;
  for (int kb = 0; kb < kIters; ++kb) {
    __syncthreads();                  // protect LDS from prior iter's readers
#pragma unroll
    for (int i = 0; i < 2; ++i) {
      int c = tid + i * 128;          // 256 16B chunks per tile
      int r = c >> 2, cc = c & 3;
      async_ld_b128(lds_off32(&At[r * 40 + cc * 8]),
                    Ag + (size_t)(bm + r) * Kdim + kb * 32 + cc * 8);
      async_ld_b128(lds_off32(&Bt[r * 40 + cc * 8]),
                    Bg + (size_t)(bn + r) * Kdim + kb * 32 + cc * 8);
    }
    if (kb + 1 < kIters) {
      __builtin_prefetch(Ag + (size_t)(bm + (tid >> 1)) * Kdim + (kb + 1) * 32, 0, 0);
      __builtin_prefetch(Bg + (size_t)(bn + (tid >> 1)) * Kdim + (kb + 1) * 32, 0, 0);
    }
    wait_async0();                    // own async copies landed in LDS
    __syncthreads();                  // all waves' copies visible

    v16bf aF = load_a_frag(&At[wave * 16 * 40], 40);
#pragma unroll
    for (int j = 0; j < 4; ++j) {
      v16bf bF = load_b_frag(&Bt[j * 16 * 40], 40);
      acc[j] = wmma_bf16(aF, bF, acc[j]);
    }
  }

  // Epilogue. C/D layout: lane = N (mod 16), VGPR i = row M, upper lanes M+8.
  const int nLoc  = lane & 15;
  const int mHalf = lane >> 4;
#pragma unroll
  for (int j = 0; j < 4; ++j) {
    int cg = bn + j * 16 + nLoc;
    float bv = bias[cg];
#pragma unroll
    for (int i = 0; i < 8; ++i) {
      int rg = bm + wave * 16 + mHalf * 8 + i;
      float val = acc[j][i] + bv;
      if (mode == 0) {
        outF[(size_t)rg * Ndim + cg] = val;
      } else {
        int sec = cg >> 10, rem = cg & 1023;
        int hh = rem >> 6, dd = rem & 63;
        int b = rg >> 11, t = rg & 2047;
        int bh = b * NHEAD + hh;
        __bf16 bvb = f2bf(val);
        if (sec == 0)      qO[((size_t)bh * SEQ + t) * HEADD + dd] = bvb;
        else if (sec == 1) kO[((size_t)bh * SEQ + t) * HEADD + dd] = bvb;
        else               vTO[((size_t)bh * HEADD + dd) * SEQ + t] = bvb;  // V transposed
      }
    }
  }
}

// Flash attention: per block one (b,h), 64 queries (16/wave), stream 32-key tiles.
__global__ void __launch_bounds__(128)
flash_attn_bf16(const __bf16* __restrict__ qg, const __bf16* __restrict__ kg,
                const __bf16* __restrict__ vT, __bf16* __restrict__ y) {
  __shared__ __bf16 Kt[32 * 72];        // [key][d]  (d contiguous, ld 72)
  __shared__ __bf16 Vt[64 * 40];        // [d][key]  (key contiguous, ld 40)
  __shared__ __bf16 Pt[4][16 * 40];     // per-wave P re-layout scratch

  const int tid  = threadIdx.x;
  const int wave = tid >> 5;
  const int lane = tid & 31;
  const int bh = blockIdx.y;            // b*16+h
  const int b = bh >> 4, h = bh & 15;
  const int q0 = blockIdx.x * 64 + wave * 16;

  const float scl = 0.125f * 1.44269504088896340736f;  // 1/sqrt(D) * log2(e)

  // Q fragments held in registers across entire key loop
  const __bf16* qbase = qg + ((size_t)bh * SEQ + q0) * HEADD;
  v16bf aQ0 = load_a_frag(qbase, HEADD);        // d 0..31
  v16bf aQ1 = load_a_frag(qbase + 32, HEADD);   // d 32..63

  float mrow[8], lrow[8];
#pragma unroll
  for (int i = 0; i < 8; ++i) { mrow[i] = -1e30f; lrow[i] = 0.f; }
  v8f o[4] = {};

  for (int kt = 0; kt < SEQ / 32; ++kt) {
    const int kbase = kt * 32;
    __syncthreads();
    {
      int r = tid >> 2, cc = tid & 3;   // 32 rows x 4 chunks (K tile)
      async_ld_b128(lds_off32(&Kt[r * 72 + cc * 8]),
                    kg + ((size_t)bh * SEQ + kbase + r) * HEADD + cc * 8);
      int rv = tid >> 1, cv = tid & 1;  // 64 rows x 2 chunks (V tile)
      async_ld_b128(lds_off32(&Vt[rv * 40 + cv * 8]),
                    vT + ((size_t)bh * HEADD + rv) * SEQ + kbase + cv * 8);
    }
    wait_async0();
    __syncthreads();

    // S = Q * K^T : two 16-key groups, K-dim (=d) chunks of 32
    v8f s0 = {}, s1 = {};
    s0 = wmma_bf16(aQ0, load_b_frag(&Kt[0], 72), s0);
    s0 = wmma_bf16(aQ1, load_b_frag(&Kt[32], 72), s0);
    s1 = wmma_bf16(aQ0, load_b_frag(&Kt[16 * 72], 72), s1);
    s1 = wmma_bf16(aQ1, load_b_frag(&Kt[16 * 72 + 32], 72), s1);
    s0 *= scl; s1 *= scl;               // log2-domain logits

    __bf16* Pw = &Pt[wave][0];
#pragma unroll
    for (int i = 0; i < 8; ++i) {       // row m = (lane>>4)*8 + i
      float mx = fmaxf(s0[i], s1[i]);
#pragma unroll
      for (int d_ = 1; d_ < 16; d_ <<= 1) mx = fmaxf(mx, __shfl_xor(mx, d_, 32));
      float mnew  = fmaxf(mrow[i], mx);
      float alpha = exp2f(mrow[i] - mnew);
      float p0 = exp2f(s0[i] - mnew);
      float p1 = exp2f(s1[i] - mnew);
      float rs = p0 + p1;
#pragma unroll
      for (int d_ = 1; d_ < 16; d_ <<= 1) rs += __shfl_xor(rs, d_, 32);
      lrow[i] = lrow[i] * alpha + rs;
      mrow[i] = mnew;
#pragma unroll
      for (int j = 0; j < 4; ++j) o[j][i] *= alpha;
      int m = (lane >> 4) * 8 + i;      // C-layout -> A-layout via LDS
      Pw[m * 40 + (lane & 15)]      = f2bf(p0);
      Pw[m * 40 + 16 + (lane & 15)] = f2bf(p1);
    }

    v16bf aP = load_a_frag(Pw, 40);     // P as 16x32 A fragment
#pragma unroll
    for (int j = 0; j < 4; ++j)
      o[j] = wmma_bf16(aP, load_b_frag(&Vt[j * 16 * 40], 40), o[j]);
  }

  // O /= l, write y as bf16 [B,T,C]
#pragma unroll
  for (int i = 0; i < 8; ++i) {
    float inv = 1.0f / lrow[i];
    int t = q0 + (lane >> 4) * 8 + i;
#pragma unroll
    for (int j = 0; j < 4; ++j) {
      int dd = j * 16 + (lane & 15);
      y[((size_t)b * SEQ + t) * EMB + h * HEADD + dd] = f2bf(o[j][i] * inv);
    }
  }
}

extern "C" void kernel_launch(void* const* d_in, const int* in_sizes, int n_in,
                              void* d_out, int out_size, void* d_ws, size_t ws_size,
                              hipStream_t stream) {
  const float* x      = (const float*)d_in[0];
  const float* W_attn = (const float*)d_in[1];
  const float* b_attn = (const float*)d_in[2];
  const float* W_proj = (const float*)d_in[3];
  const float* b_proj = (const float*)d_in[4];
  float* out = (float*)d_out;

  char* ws = (char*)d_ws;                       // 48 MB used
  __bf16* xb  = (__bf16*)(ws);                  // x bf16        [4096,1024]  8 MB
  __bf16* Wab = (__bf16*)(ws + (8u  << 20));    // W_attn bf16   [3072,1024]  6 MB
  __bf16* Wpb = (__bf16*)(ws + (14u << 20));    // W_proj bf16   [1024,1024]  2 MB
  __bf16* qB  = (__bf16*)(ws + (16u << 20));    // q [B,H,T,D]               8 MB
  __bf16* kB  = (__bf16*)(ws + (24u << 20));    // k [B,H,T,D]               8 MB
  __bf16* vTB = (__bf16*)(ws + (32u << 20));    // v [B,H,D,T]               8 MB
  __bf16* yB  = (__bf16*)(ws + (40u << 20));    // attn out bf16 [4096,1024] 8 MB

  cvt_f32_bf16<<<(MTOK * EMB) / 256, 256, 0, stream>>>(x, xb, MTOK * EMB);
  cvt_f32_bf16<<<(N_QKV * EMB) / 256, 256, 0, stream>>>(W_attn, Wab, N_QKV * EMB);
  cvt_f32_bf16<<<(EMB * EMB) / 256, 256, 0, stream>>>(W_proj, Wpb, EMB * EMB);

  // QKV projection: [4096,1024] x [3072,1024]^T, scatter epilogue
  gemm_bf16_nt<<<dim3(N_QKV / 64, MTOK / 64), 128, 0, stream>>>(
      xb, Wab, b_attn, nullptr, qB, kB, vTB, N_QKV, EMB, 1);

  // Flash attention over all (b,h), 64-query blocks
  flash_attn_bf16<<<dim3(SEQ / 64, BSZ * NHEAD), 128, 0, stream>>>(qB, kB, vTB, yB);

  // Output projection: [4096,1024] x [1024,1024]^T -> fp32 out
  gemm_bf16_nt<<<dim3(EMB / 64, MTOK / 64), 128, 0, stream>>>(
      yB, Wpb, b_proj, out, nullptr, nullptr, nullptr, EMB, EMB, 0);
}
